// Segmentation_Swin_Encoder_5978594476691
// MI455X (gfx1250) — compile-verified, matrix operand order checked by reading, not run on hardware
//
#include <hip/hip_runtime.h>
#include <hip/hip_bf16.h>
#include <math.h>

typedef __attribute__((ext_vector_type(16))) _Float16 v16h;
typedef __attribute__((ext_vector_type(8)))  _Float16 v8h;
typedef __attribute__((ext_vector_type(8)))  float    v8f;

#define WS   7
#define NW   343          // tokens per window (and number of windows)
#define NWP  352          // padded to 22*16
#define CH   16
#define NH   2
#define HD   8
#define G    49           // token grid per dim
#define NTOK (G * G * G)  // 117649
#define MLPH 64

static __device__ __forceinline__ v8f wmma16x16x32(v16h a, v16h b, v8f c) {
  return __builtin_amdgcn_wmma_f32_16x16x32_f16(false, a, false, b, (short)0, c,
                                                false, false);
}

// Per ISA 7.12.2, a 16-bit A/B fragment for lane l is two contiguous K-runs:
// [kbase, kbase+8) in vgpr0..3 and [16+kbase, 16+kbase+8) in vgpr4..7,
// kbase = (l<16) ? 0 : 8.  Build fragments from one/two aligned v8h LDS loads.
static __device__ __forceinline__ v16h frag_lo(v8h lo) {
  v16h r = {};
#pragma unroll
  for (int i = 0; i < 8; i++) r[i] = lo[i];
  return r;
}
static __device__ __forceinline__ v16h frag_pair(v8h lo, v8h hi) {
  v16h r;
#pragma unroll
  for (int i = 0; i < 8; i++) { r[i] = lo[i]; r[i + 8] = hi[i]; }
  return r;
}
static __device__ __forceinline__ v8h ldsv8(const _Float16* p) {
  return *(const v8h*)p;  // aligned ds_load_b128
}

// 16-lane (half-wave / DPP row) reductions via DPP modifiers: pure VALU,
// no ds_bpermute, no DScnt waits.  Pairing sequence i^1, i^2, i^7, i^15
// covers all 16 lanes of each DPP row.
template <int CTRL>
static __device__ __forceinline__ float dpp_movf(float x) {
  return __int_as_float(__builtin_amdgcn_update_dpp(
      0, __float_as_int(x), CTRL, 0xf, 0xf, true));
}
static __device__ __forceinline__ float red16_max(float x) {
  x = fmaxf(x, dpp_movf<0xB1>(x));    // quad_perm {1,0,3,2}  : i^1
  x = fmaxf(x, dpp_movf<0x4E>(x));    // quad_perm {2,3,0,1}  : i^2
  x = fmaxf(x, dpp_movf<0x141>(x));   // row_half_mirror      : i^7
  x = fmaxf(x, dpp_movf<0x140>(x));   // row_mirror           : i^15
  return x;
}
static __device__ __forceinline__ float red16_sum(float x) {
  x += dpp_movf<0xB1>(x);
  x += dpp_movf<0x4E>(x);
  x += dpp_movf<0x141>(x);
  x += dpp_movf<0x140>(x);
  return x;
}

// ---------------------------------------------------------------------------
// Patch embed: 2x2x2 stride-2 conv (1->16) + bias + LayerNorm over C
// ---------------------------------------------------------------------------
__global__ __launch_bounds__(256) void patch_embed_kernel(
    const float* __restrict__ x, const float* __restrict__ pe_w,
    const float* __restrict__ pe_bias, const float* __restrict__ pe_g,
    const float* __restrict__ pe_b, float* __restrict__ act) {
  __shared__ float w[16 * 8];
  __shared__ float bias[16], g[16], bb[16];
  const int tid = threadIdx.x;
  if (tid < 128) w[tid] = pe_w[tid];
  if (tid < 16) { bias[tid] = pe_bias[tid]; g[tid] = pe_g[tid]; bb[tid] = pe_b[tid]; }
  __syncthreads();
  int t = blockIdx.x * 256 + tid;
  if (t >= NTOK) return;
  const int IN = 98;
  int od = t / (G * G), oh = (t / G) % G, ow = t % G;
  float in[8];
#pragma unroll
  for (int k = 0; k < 8; k++) {
    int kd = k >> 2, kh = (k >> 1) & 1, kw = k & 1;
    in[k] = x[(size_t)((od * 2 + kd) * IN + (oh * 2 + kh)) * IN + (ow * 2 + kw)];
  }
  float h[16];
  float mu = 0.f;
#pragma unroll
  for (int c = 0; c < 16; c++) {
    float a = bias[c];
#pragma unroll
    for (int k = 0; k < 8; k++) a += in[k] * w[c * 8 + k];
    h[c] = a;
    mu += a;
  }
  mu *= (1.f / 16.f);
  float var = 0.f;
#pragma unroll
  for (int c = 0; c < 16; c++) { float d = h[c] - mu; var += d * d; }
  float rs = rsqrtf(var * (1.f / 16.f) + 1e-5f);
#pragma unroll
  for (int c = 0; c < 16; c++)
    act[(size_t)t * CH + c] = (h[c] - mu) * rs * g[c] + bb[c];
}

// ---------------------------------------------------------------------------
// Dense padded relative-position-bias table (window independent):
// bias_tab[head][m(352)][n(352)]; padded columns get -1e30 (doubles as the
// softmax column mask).  Recomputed once per Swin block (~248K elements).
// ---------------------------------------------------------------------------
__global__ __launch_bounds__(256) void bias_precompute_kernel(
    const float* __restrict__ rpb, float* __restrict__ bias_tab) {
  int i = blockIdx.x * 256 + threadIdx.x;
  const int TOT = NH * NWP * NWP;
  if (i >= TOT) return;
  int n = i % NWP;
  int m = (i / NWP) % NWP;
  int head = i / (NWP * NWP);
  float v = -1e30f;
  if (m < NW && n < NW) {
    int mz = m / 49, my = (m / 7) % 7, mx = m % 7;
    int nz = n / 49, ny = (n / 7) % 7, nx = n % 7;
    int idx = ((mz - nz + 6) * 13 + (my - ny + 6)) * 13 + (mx - nx + 6);
    v = rpb[(size_t)idx * NH + head];
  }
  bias_tab[i] = v;
}

// ---------------------------------------------------------------------------
// Fused Swin attention block: one workgroup per 7x7x7 window.
// LN1 + QKV (VALU), flash attention with WMMA, proj + residual.
// ---------------------------------------------------------------------------
__global__ __launch_bounds__(256) void swin_attn_kernel(
    float* __restrict__ act, const float* __restrict__ qkv_w,
    const float* __restrict__ n1_g, const float* __restrict__ n1_b,
    const float* __restrict__ bias_tab,  // (NH, NWP, NWP) padded table
    const float* __restrict__ proj_w, const float* __restrict__ proj_b) {
  __shared__ alignas(16) _Float16 qh[NWP][CH];   // q*scale, [token][head*8+d]
  __shared__ alignas(16) _Float16 kh[NWP][CH];   // [token][head*8+d]
  __shared__ alignas(16) _Float16 vt[CH][NWP];   // transposed: [head*8+d][token]
  __shared__ alignas(16) _Float16 obuf[NWP][CH];
  __shared__ alignas(16) _Float16 pstage[8][16][16];  // per-wave P staging
  __shared__ float wq[48 * 16];
  __shared__ float wproj[16 * 16];
  __shared__ float sg[CH], sb[CH], spb[CH];

  const int tid = threadIdx.x;
  const int wid = tid >> 5;
  const int lane = tid & 31;
  const int win = blockIdx.x;
  const int wd = win / 49, wh = (win / 7) % 7, ww = win % 7;

  __builtin_prefetch(bias_tab, 0, 1);  // global_prefetch_b8 of hot bias table

  for (int i = tid; i < 48 * 16; i += 256) wq[i] = qkv_w[i];
  for (int i = tid; i < 16 * 16; i += 256) wproj[i] = proj_w[i];
  if (tid < CH) { sg[tid] = n1_g[tid]; sb[tid] = n1_b[tid]; spb[tid] = proj_b[tid]; }
  __syncthreads();

  const float scale = 0.35355339059327373f;  // 8^-0.5

  // ---- phase 1: LN1 + QKV projection per token ----
  for (int t = tid; t < NWP; t += 256) {
    if (t < NW) {
      int td = t / 49, th = (t / 7) % 7, tw = t % 7;
      int gd = wd * 7 + td, gh = wh * 7 + th, gw = ww * 7 + tw;
      const float* xp = act + (size_t)((gd * G + gh) * G + gw) * CH;
      float xv[CH], y[CH];
      float mu = 0.f;
#pragma unroll
      for (int c = 0; c < CH; c++) { xv[c] = xp[c]; mu += xv[c]; }
      mu *= (1.f / CH);
      float var = 0.f;
#pragma unroll
      for (int c = 0; c < CH; c++) { float d = xv[c] - mu; var += d * d; }
      float rs = rsqrtf(var * (1.f / CH) + 1e-5f);
#pragma unroll
      for (int c = 0; c < CH; c++) y[c] = (xv[c] - mu) * rs * sg[c] + sb[c];
      for (int j = 0; j < CH; j++) {
        float aq = 0.f, ak = 0.f, av = 0.f;
#pragma unroll
        for (int c = 0; c < CH; c++) {
          aq += y[c] * wq[j * CH + c];
          ak += y[c] * wq[(16 + j) * CH + c];
          av += y[c] * wq[(32 + j) * CH + c];
        }
        qh[t][j] = (_Float16)(aq * scale);
        kh[t][j] = (_Float16)ak;
        vt[j][t] = (_Float16)av;
      }
    } else {
      for (int j = 0; j < CH; j++) {
        qh[t][j] = (_Float16)0.f;
        kh[t][j] = (_Float16)0.f;
        vt[j][t] = (_Float16)0.f;
      }
    }
  }
  __syncthreads();

  // ---- phase 2: flash attention, one 16-row M tile per wave iteration ----
  const int half = lane >> 4;        // 0: kbase=0 (real K), 1: kbase=8 (pad)
  const int l15 = lane & 15;
  const int kbase = half * 8;
  const v8h zero8 = {};

  for (int mt = wid; mt < 22; mt += 8) {
    const int m0 = mt * 16;
    for (int head = 0; head < NH; head++) {
      // Q fragment: A 16x32, real K = 8. One b128 load + cndmask, no branch.
      v8h qd = ldsv8(&qh[m0 + l15][head * HD]);
      v16h qa = frag_lo(half == 0 ? qd : zero8);

      // hoisted bias row base: rows m0+half*8+v, column l15 (+n0 per tile)
      const float* bptr = bias_tab + ((size_t)head * NWP + (m0 + half * 8)) * NWP + l15;

      v8f oacc;
      float mrow[8], lrow[8];
#pragma unroll
      for (int v = 0; v < 8; v++) { oacc[v] = 0.f; mrow[v] = -1e30f; lrow[v] = 0.f; }

#pragma unroll 1
      for (int nt = 0; nt < 22; nt++) {
        const int n0 = nt * 16;
        // K fragment: B 32x16, real K = 8
        v8h kd = ldsv8(&kh[n0 + l15][head * HD]);
        v16h kb = frag_lo(half == 0 ? kd : zero8);

        v8f s;
#pragma unroll
        for (int v = 0; v < 8; v++) s[v] = 0.f;
        s = wmma16x16x32(qa, kb, s);  // S tile = q.k^T (scale folded into q)

        // bias + column mask: 8 unconditional coalesced loads (L2-resident)
#pragma unroll
        for (int v = 0; v < 8; v++) s[v] += bptr[(size_t)v * NWP + n0];

        // online softmax: DPP16 row reductions (no LDS, no DScnt waits)
#pragma unroll
        for (int v = 0; v < 8; v++) {
          float tm = red16_max(s[v]);
          float nm = fmaxf(mrow[v], tm);
          float corr = __expf(mrow[v] - nm);
          mrow[v] = nm;
          float p = __expf(s[v] - nm);
          s[v] = p;
          float ts = red16_sum(p);
          lrow[v] = lrow[v] * corr + ts;
          oacc[v] *= corr;
        }
        // stage P (C-layout) -> LDS, re-read in A-fragment layout
#pragma unroll
        for (int v = 0; v < 8; v++)
          pstage[wid][v + half * 8][l15] = (_Float16)s[v];
        asm volatile("s_wait_dscnt 0" ::: "memory");  // same-wave DS RAW

        // P fragment: A 16x32, real K = 16 -> unconditional b128, no select
        v16h pa = frag_lo(ldsv8(&pstage[wid][l15][kbase]));
        // V fragment: B 32x16 from transposed V, run contiguous along tokens
        v8h vd = ldsv8(&vt[head * HD + (l15 & 7)][n0 + kbase]);
        v16h vb = frag_lo(l15 < HD ? vd : zero8);

        oacc = wmma16x16x32(pa, vb, oacc);
      }
      // finalize rows of this head
#pragma unroll
      for (int v = 0; v < 8; v++) {
        int mloc = m0 + v + half * 8;
        if (mloc < NW && l15 < HD)
          obuf[mloc][head * HD + l15] = (_Float16)(oacc[v] / lrow[v]);
      }
    }
  }
  __syncthreads();

  // ---- phase 3: output projection + residual ----
  for (int t = tid; t < NW; t += 256) {
    int td = t / 49, th = (t / 7) % 7, tw = t % 7;
    int gd = wd * 7 + td, gh = wh * 7 + th, gw = ww * 7 + tw;
    float* xp = act + (size_t)((gd * G + gh) * G + gw) * CH;
    float o[CH];
#pragma unroll
    for (int c = 0; c < CH; c++) o[c] = (float)obuf[t][c];
#pragma unroll
    for (int j = 0; j < CH; j++) {
      float a = spb[j];
#pragma unroll
      for (int c = 0; c < CH; c++) a += o[c] * wproj[j * CH + c];
      xp[j] = xp[j] + a;
    }
  }
}

// ---------------------------------------------------------------------------
// Fused MLP block: LN2 + fc1 + exact GELU + fc2 + residual. One 16-token
// tile per wave; both GEMMs on WMMA (K padded to 32 / split 64 = 2x32).
// Weight tiles staged B-transposed so fragment runs are contiguous.
// ---------------------------------------------------------------------------
__global__ __launch_bounds__(256) void swin_mlp_kernel(
    float* __restrict__ act, const float* __restrict__ n2_g,
    const float* __restrict__ n2_b, const float* __restrict__ fc1_w,
    const float* __restrict__ fc1_b, const float* __restrict__ fc2_w,
    const float* __restrict__ fc2_b) {
  __shared__ alignas(16) _Float16 w1t[MLPH][CH];  // [n=hidden][k=in ch]
  __shared__ alignas(16) _Float16 w2t[CH][MLPH];  // [n=out ch][k=hidden]
  __shared__ float b1[MLPH], b2[CH], g2[CH], bb2[CH];
  __shared__ alignas(16) float    xtile[8][16][CH];
  __shared__ alignas(16) _Float16 ztile[8][16][CH];
  __shared__ alignas(16) _Float16 h1tile[8][16][MLPH];

  const int tid = threadIdx.x, wid = tid >> 5, lane = tid & 31;
  // w1t == natural row-major of fc1_w (64,16); w2t == row-major fc2_w (16,64)
  for (int i = tid; i < MLPH * CH; i += 256)
    ((_Float16*)w1t)[i] = (_Float16)fc1_w[i];
  for (int i = tid; i < CH * MLPH; i += 256)
    ((_Float16*)w2t)[i] = (_Float16)fc2_w[i];
  if (tid < MLPH) b1[tid] = fc1_b[tid];
  if (tid < CH) { b2[tid] = fc2_b[tid]; g2[tid] = n2_g[tid]; bb2[tid] = n2_b[tid]; }
  __syncthreads();

  const int half = lane >> 4, l15 = lane & 15, kbase = half * 8;
  const int t0 = (blockIdx.x * 8 + wid) * 16;
  if (t0 >= NTOK) return;  // wave-uniform; no barriers past this point

  // phase 1: load + LN2 (each lane: token l15, channels kbase..kbase+7)
  const int t = t0 + l15;
  const bool valid = (t < NTOK);
  float xv[8], psum = 0.f;
#pragma unroll
  for (int c = 0; c < 8; c++) {
    float x = valid ? act[(size_t)t * CH + kbase + c] : 0.f;
    xv[c] = x;
    psum += x;
  }
  float mu = (psum + __shfl_xor(psum, 16)) * (1.f / CH);
  float vs = 0.f;
#pragma unroll
  for (int c = 0; c < 8; c++) { float d = xv[c] - mu; vs += d * d; }
  float rs = rsqrtf((vs + __shfl_xor(vs, 16)) * (1.f / CH) + 1e-5f);
#pragma unroll
  for (int c = 0; c < 8; c++) {
    float z = (xv[c] - mu) * rs * g2[kbase + c] + bb2[kbase + c];
    ztile[wid][l15][kbase + c] = (_Float16)(valid ? z : 0.f);
    xtile[wid][l15][kbase + c] = xv[c];
  }
  asm volatile("s_wait_dscnt 0" ::: "memory");

  // phase 2: fc1 (K=16 padded to 32, N=64 as 4 tiles) + exact GELU
  v16h za = frag_lo(ldsv8(&ztile[wid][l15][kbase]));
#pragma unroll
  for (int nt = 0; nt < 4; nt++) {
    v16h wb = frag_lo(ldsv8(&w1t[nt * 16 + l15][kbase]));
    v8f acc;
    float bn = b1[nt * 16 + l15];
#pragma unroll
    for (int v = 0; v < 8; v++) acc[v] = bn;
    acc = wmma16x16x32(za, wb, acc);
#pragma unroll
    for (int v = 0; v < 8; v++) {
      float xg = acc[v];
      float ge = 0.5f * xg * (1.f + erff(xg * 0.70710678118654752f));
      h1tile[wid][v + half * 8][nt * 16 + l15] = (_Float16)ge;
    }
  }
  asm volatile("s_wait_dscnt 0" ::: "memory");

  // phase 3: fc2 (K=64 as two chained 32-K WMMAs), + bias + residual
  v8f acc2;
  float bc = b2[l15];
#pragma unroll
  for (int v = 0; v < 8; v++) acc2[v] = bc;
#pragma unroll
  for (int kc = 0; kc < 2; kc++) {
    v16h ha = frag_pair(ldsv8(&h1tile[wid][l15][kc * 32 + kbase]),
                        ldsv8(&h1tile[wid][l15][kc * 32 + 16 + kbase]));
    v16h wb2 = frag_pair(ldsv8(&w2t[l15][kc * 32 + kbase]),
                         ldsv8(&w2t[l15][kc * 32 + 16 + kbase]));
    acc2 = wmma16x16x32(ha, wb2, acc2);
  }
#pragma unroll
  for (int v = 0; v < 8; v++) {
    int mm = v + half * 8;
    int tt = t0 + mm;
    if (tt < NTOK)
      act[(size_t)tt * CH + l15] = xtile[wid][mm][l15] + acc2[v];
  }
}

// ---------------------------------------------------------------------------
// (token, C) -> (C, token) for the final (b, c, d, h, w) output
// ---------------------------------------------------------------------------
__global__ __launch_bounds__(256) void transpose_out_kernel(
    const float* __restrict__ act, float* __restrict__ out) {
  int i = blockIdx.x * 256 + threadIdx.x;
  if (i >= NTOK * CH) return;
  int tk = i / CH, c = i % CH;
  out[(size_t)c * NTOK + tk] = act[i];
}

extern "C" void kernel_launch(void* const* d_in, const int* in_sizes, int n_in,
                              void* d_out, int out_size, void* d_ws,
                              size_t ws_size, hipStream_t stream) {
  (void)in_sizes; (void)n_in; (void)out_size; (void)ws_size;
  const float* x       = (const float*)d_in[0];
  const float* pe_w    = (const float*)d_in[1];
  const float* pe_bias = (const float*)d_in[2];
  const float* pe_g    = (const float*)d_in[3];
  const float* pe_b    = (const float*)d_in[4];
  const float* n1_g    = (const float*)d_in[5];
  const float* n1_b    = (const float*)d_in[6];
  const float* qkv_w   = (const float*)d_in[7];
  const float* rpb     = (const float*)d_in[8];
  const float* proj_w  = (const float*)d_in[9];
  const float* proj_b  = (const float*)d_in[10];
  const float* n2_g    = (const float*)d_in[11];
  const float* n2_b    = (const float*)d_in[12];
  const float* fc1_w   = (const float*)d_in[13];
  const float* fc1_b   = (const float*)d_in[14];
  const float* fc2_w   = (const float*)d_in[15];
  const float* fc2_b   = (const float*)d_in[16];

  float* act = (float*)d_ws;            // 117649 x 16 fp32
  float* bias_tab = act + (size_t)NTOK * CH;  // NH x 352 x 352 fp32 (~991 KB)

  patch_embed_kernel<<<(NTOK + 255) / 256, 256, 0, stream>>>(
      x, pe_w, pe_bias, pe_g, pe_b, act);

  const int bias_elems = NH * NWP * NWP;
  const int mlp_tiles = (NTOK + 15) / 16;          // 7354
  const int mlp_blocks = (mlp_tiles + 7) / 8;      // 920
  for (int i = 0; i < 3; i++) {
    bias_precompute_kernel<<<(bias_elems + 255) / 256, 256, 0, stream>>>(
        rpb + (size_t)i * 2197 * NH, bias_tab);
    swin_attn_kernel<<<NW, 256, 0, stream>>>(
        act, qkv_w + (size_t)i * 48 * 16, n1_g + i * 16, n1_b + i * 16,
        bias_tab, proj_w + (size_t)i * 256, proj_b + i * 16);
    swin_mlp_kernel<<<mlp_blocks, 256, 0, stream>>>(
        act, n2_g + i * 16, n2_b + i * 16, fc1_w + (size_t)i * MLPH * CH,
        fc1_b + i * MLPH, fc2_w + (size_t)i * CH * MLPH, fc2_b + i * 16);
  }

  transpose_out_kernel<<<(NTOK * CH + 255) / 256, 256, 0, stream>>>(
      act, (float*)d_out);
}
